// CenterAwarePseudoModule_37065567764815
// MI455X (gfx1250) — compile-verified, matrix operand order checked by reading, not run on hardware
//
#include <hip/hip_runtime.h>
#include <hip/hip_bf16.h>

typedef __attribute__((ext_vector_type(16))) _Float16 v16h;
typedef __attribute__((ext_vector_type(8)))  _Float16 v8h;
typedef __attribute__((ext_vector_type(8)))  float    v8f;

// Problem constants (from reference): N=262144, D=512, K=128
static constexpr int DIM     = 512;   // feature dim used in GEMM (D)
static constexpr int NCENT   = 128;   // K centroids
static constexpr int PITCH   = 520;   // f16 elements per centroid row in LDS/ws (pad: 1040B -> bank stride 4)
static constexpr int CENT_BYTES = NCENT * PITCH * 2;          // 133120
static constexpr int COPY_BYTES = CENT_BYTES + 2 * NCENT * 4; // + bsq + bias = 134144 (16B multiple)
static constexpr int SMEM_BYTES = COPY_BYTES + 128 * 4;       // + per-wave invnorm area = 134656

// ---------------------------------------------------------------------------
// Prep: convert centroids to f16 (padded layout), compute |c|^2 and bias.
// One block per centroid.
// ---------------------------------------------------------------------------
__global__ __launch_bounds__(256) void nc_prep(const float* __restrict__ initc,
                                               _Float16* __restrict__ hc,
                                               float* __restrict__ bsq,
                                               float* __restrict__ bias)
{
    __shared__ float red[256];
    const int b   = blockIdx.x;
    const int tid = threadIdx.x;
    const float* crow = initc + (size_t)b * (DIM + 1);

    float ss = 0.0f;
    for (int k = tid; k < DIM + 1; k += 256) {
        float v = crow[k];
        ss += v * v;
        if (k < DIM) hc[(size_t)b * PITCH + k] = (_Float16)v;
    }
    red[tid] = ss;
    __syncthreads();
    for (int s = 128; s > 0; s >>= 1) {
        if (tid < s) red[tid] += red[tid + s];
        __syncthreads();
    }
    if (tid == 0) {
        bsq[b]  = red[0];
        bias[b] = crow[DIM];
    }
}

// ---------------------------------------------------------------------------
// Main: fused GEMM (f16 WMMA, f32 acc) + row-norm + argmin.
// 256 threads = 8 waves; each wave: 16 rows x 128 centroids.
// All B-fragment ds_loads use immediate offsets off two loop-carried bases so
// the inner loop has no VALU address math (no WMMA->VALU WAR nops).
// ---------------------------------------------------------------------------
__global__ __launch_bounds__(256) void nc_main(const float* __restrict__ feats,
                                               const void* __restrict__ wsbase,
                                               const int* __restrict__ labels,
                                               float* __restrict__ out)
{
    extern __shared__ char smem[];
    _Float16* sc    = (_Float16*)smem;                 // [NCENT][PITCH] f16 centroids
    float*    sbsq  = (float*)(smem + CENT_BYTES);     // [NCENT] |c|^2
    float*    sbias = sbsq + NCENT;                    // [NCENT] bias (c[512])
    float*    sinv  = sbias + NCENT;                   // [8 waves][16 rows] 1/||f||

    const int tid = threadIdx.x;

    // Cooperative flat copy ws -> LDS (hot in L2 after prep; 134144 B as uint4)
    {
        const uint4* src = (const uint4*)wsbase;
        uint4*       dst = (uint4*)smem;
        for (int i = tid; i < COPY_BYTES / 16; i += 256) dst[i] = src[i];
    }
    __syncthreads();

    const int wave = tid >> 5;
    const int lane = tid & 31;
    const int m16  = lane & 15;   // A row / B column within tile
    const int g    = lane >> 4;   // lane group: selects K sub-range per ISA layout

    const int rowBase = blockIdx.x * 128 + wave * 16;
    const float* frow = feats + (size_t)(rowBase + m16) * DIM;

    v8f acc[8];
#pragma unroll
    for (int t = 0; t < 8; ++t) acc[t] = (v8f){};

    float ssq = 0.0f;

    // Loop-carried LDS base pointers: p0 -> tiles 0..3, p1 -> tiles 4..7.
    // Tile strides within each group are compile-time DS immediate offsets
    // (max 3*16640+32 = 49952 < 65536).
    const _Float16* p0 = sc + (size_t)m16 * PITCH + 8 * g;
    const _Float16* p1 = p0 + 64 * PITCH;

    // Prologue: A chunk 0 in flight.
    float4 c0 = *(const float4*)(frow + 8 * g);
    float4 c1 = *(const float4*)(frow + 8 * g + 4);
    float4 c2 = *(const float4*)(frow + 16 + 8 * g);
    float4 c3 = *(const float4*)(frow + 16 + 8 * g + 4);

    for (int kc = 0; kc < DIM / 32; ++kc) {
        // Issue ALL 16 ds_loads for this chunk's 8 B fragments up front with
        // immediate offsets; no address VALU between the WMMAs below.
        v16h Bf[8];
#pragma unroll
        for (int t = 0; t < 4; ++t) {
            v8h blo = *(const v8h*)(p0 + t * (16 * PITCH));
            v8h bhi = *(const v8h*)(p0 + t * (16 * PITCH) + 16);
#pragma unroll
            for (int i = 0; i < 8; ++i) { Bf[t][i] = blo[i]; Bf[t][i + 8] = bhi[i]; }
        }
#pragma unroll
        for (int t = 0; t < 4; ++t) {
            v8h blo = *(const v8h*)(p1 + t * (16 * PITCH));
            v8h bhi = *(const v8h*)(p1 + t * (16 * PITCH) + 16);
#pragma unroll
            for (int i = 0; i < 8; ++i) { Bf[t + 4][i] = blo[i]; Bf[t + 4][i + 8] = bhi[i]; }
        }

        // Double-buffer: next chunk's A global loads issued before this
        // chunk's WMMAs (uniform wrap to 0 on last iter; result discarded).
        const int kb  = kc * 32;
        const int kbn = (kc + 1 < DIM / 32) ? kb + 32 : 0;
        float4 n0 = *(const float4*)(frow + kbn + 8 * g);
        float4 n1 = *(const float4*)(frow + kbn + 8 * g + 4);
        float4 n2 = *(const float4*)(frow + kbn + 16 + 8 * g);
        float4 n3 = *(const float4*)(frow + kbn + 16 + 8 * g + 4);
        __builtin_prefetch(frow + kbn + 32 + 8 * g, 0, 3);  // chunk kc+2, near-scope

        ssq += c0.x*c0.x + c0.y*c0.y + c0.z*c0.z + c0.w*c0.w
             + c1.x*c1.x + c1.y*c1.y + c1.z*c1.z + c1.w*c1.w
             + c2.x*c2.x + c2.y*c2.y + c2.z*c2.z + c2.w*c2.w
             + c3.x*c3.x + c3.y*c3.y + c3.z*c3.z + c3.w*c3.w;

        v16h A;
        A[0]=(_Float16)c0.x;  A[1]=(_Float16)c0.y;  A[2]=(_Float16)c0.z;  A[3]=(_Float16)c0.w;
        A[4]=(_Float16)c1.x;  A[5]=(_Float16)c1.y;  A[6]=(_Float16)c1.z;  A[7]=(_Float16)c1.w;
        A[8]=(_Float16)c2.x;  A[9]=(_Float16)c2.y;  A[10]=(_Float16)c2.z; A[11]=(_Float16)c2.w;
        A[12]=(_Float16)c3.x; A[13]=(_Float16)c3.y; A[14]=(_Float16)c3.z; A[15]=(_Float16)c3.w;

#pragma unroll
        for (int t = 0; t < 8; ++t) {
            acc[t] = __builtin_amdgcn_wmma_f32_16x16x32_f16(
                false, A, false, Bf[t], (short)0, acc[t], false, false);
        }

        c0 = n0; c1 = n1; c2 = n2; c3 = n3;
        p0 += 32;   // advance K by one chunk (32 halves = 64 B)
        p1 += 32;
    }

    // Full row |feats|^2: lane m and m+16 hold complementary K halves.
    ssq += __shfl_xor(ssq, 16, 32);
    const float invn = rsqrtf(ssq + 1.0f);   // 1/||[feats_n, 1]||
    sinv[wave * 16 + m16] = invn;
    __syncthreads();

    // Epilogue: score(m, l) = |c_l|^2 - 2*(dot + bias_l)*invnorm_m  (argmin-equivalent)
    float invr[8];
#pragma unroll
    for (int r = 0; r < 8; ++r) invr[r] = sinv[wave * 16 + 8 * g + r];

    float bestv[8];
    int   besti[8];
#pragma unroll
    for (int r = 0; r < 8; ++r) { bestv[r] = 3.4e38f; besti[r] = 0x7fffffff; }

#pragma unroll
    for (int t = 0; t < 8; ++t) {
        const int   n  = t * 16 + m16;
        const float bq = sbsq[n];
        const float bi = sbias[n];
#pragma unroll
        for (int r = 0; r < 8; ++r) {
            const float s = bq - 2.0f * (acc[t][r] + bi) * invr[r];
            if (s < bestv[r] || (s == bestv[r] && n < besti[r])) {
                bestv[r] = s; besti[r] = n;
            }
        }
    }

    // Reduce across the 16 lanes of this column group (ties -> smallest index,
    // matching jnp.argmin first-occurrence).
#pragma unroll
    for (int mask = 1; mask < 16; mask <<= 1) {
#pragma unroll
        for (int r = 0; r < 8; ++r) {
            float ov = __shfl_xor(bestv[r], mask, 32);
            int   oi = __shfl_xor(besti[r], mask, 32);
            if (ov < bestv[r] || (ov == bestv[r] && oi < besti[r])) {
                bestv[r] = ov; besti[r] = oi;
            }
        }
    }

    if (m16 < 8) {
        const int r   = m16;
        const int row = rowBase + 8 * g + r;
        out[row] = (float)labels[besti[r]];
    }
}

// ---------------------------------------------------------------------------
extern "C" void kernel_launch(void* const* d_in, const int* in_sizes, int n_in,
                              void* d_out, int out_size, void* d_ws, size_t ws_size,
                              hipStream_t stream)
{
    const float* feats  = (const float*)d_in[0];   // [N, 512] f32
    const float* initc  = (const float*)d_in[1];   // [128, 513] f32
    const int*   labels = (const int*)d_in[2];     // [128]
    float*       out    = (float*)d_out;           // [N] labels as float

    _Float16* hc   = (_Float16*)d_ws;
    float*    bsq  = (float*)((char*)d_ws + CENT_BYTES);
    float*    bias = bsq + NCENT;

    const int nrows  = in_sizes[0] / DIM;          // 262144
    const int blocks = nrows / 128;                // 2048

    nc_prep<<<NCENT, 256, 0, stream>>>(initc, hc, bsq, bias);
    nc_main<<<blocks, 256, SMEM_BYTES, stream>>>(feats, d_ws, labels, out);
}
